// RNNLayer_63565515981228
// MI455X (gfx1250) — matrix operand order repaired and checked by reading, not compiled, not run on hardware
//
#include <hip/hip_runtime.h>
#include <hip/hip_bf16.h>

// GRU layer fused kernel for MI455X (gfx1250, wave32, WMMA).
// B=32, L=48, N=325, D=128 -> BN=10400 rows, 3D=384 gate columns.

#define B_   32
#define L_   48
#define N_   325
#define D_   128
#define BN   10400        // B_*N_
#define G3   384          // 3*D_
#define MB   32           // rows per block (10400/32 = 325 blocks exactly)
#define HS   136          // h row stride in halves (128 + 8 pad, 16B aligned)
#define NTILES 24         // G3/16
#define KT   4            // D_/32 k-chunks per GEMM

typedef _Float16 v16h __attribute__((ext_vector_type(16)));
typedef _Float16 h8   __attribute__((ext_vector_type(8)));
typedef float    v8f  __attribute__((ext_vector_type(8)));
typedef float    f8   __attribute__((ext_vector_type(8)));

static __device__ __forceinline__ v16h cat8(h8 lo, h8 hi) {
  return __builtin_shufflevector(lo, hi, 0,1,2,3,4,5,6,7,8,9,10,11,12,13,14,15);
}

#define WMMA_F16(a,b,c) \
  __builtin_amdgcn_wmma_f32_16x16x32_f16(false,(a),false,(b),(short)0,(c),false,false)

// LDS layout (dynamic shared):
//   wiL: W_ih in B-fragment swizzle, 24 ntiles * 4 k * 2 chunks * 32 lanes * 8 halves
//   whL: W_hh same
//   hb0/hb1: double-buffered hidden state, MB rows * HS halves
#define WSWZ (NTILES*KT*2*32*8)            // 49152 halves per matrix
#define LDS_BYTES ((size_t)(2*WSWZ + 2*MB*HS) * sizeof(_Float16))  // 214,016 B

__global__ __launch_bounds__(256, 1) void gru_fused_kernel(
    const float* __restrict__ X,    // (B, L, N, D) fp32
    const float* __restrict__ Wih,  // (3D, D) fp32
    const float* __restrict__ Whh,  // (3D, D) fp32
    const float* __restrict__ bih,  // (3D,)
    const float* __restrict__ bhh,  // (3D,)
    float* __restrict__ out)        // (L, BN, D) fp32
{
  extern __shared__ char smem[];
  _Float16* wiL = (_Float16*)smem;
  _Float16* whL = wiL + WSWZ;
  _Float16* hb0 = whL + WSWZ;
  _Float16* hb1 = hb0 + MB*HS;

  const int tid  = threadIdx.x;
  const int lane = tid & 31;
  const int w    = tid >> 5;      // wave 0..7
  const int mt   = w & 1;         // M-tile within block (16 rows each)
  const int dr   = w >> 1;        // d-range 0..3 (32 cols of D each)
  const int dbase = dr * 32;
  const int row0  = blockIdx.x * MB;

  // ---- Stage weights fp32 -> f16 into LDS in B-fragment swizzled layout ----
  // Fragment entry i = ((nt*4 + k)*2 + c)*32 + lane holds 8 contiguous halves:
//   W[nt*16 + lane%16][k*32 + (lane<16?0:8) + c*16 .. +7]
  for (int i = tid; i < NTILES*KT*2*32; i += 256) {
    int ln = i & 31; int c = (i >> 5) & 1; int k = (i >> 6) & 3; int nt = i >> 8;
    int g  = nt*16 + (ln & 15);
    int kk = k*32 + ((ln < 16) ? 0 : 8) + c*16;
    f8 s_i = *(const f8*)(Wih + g*D_ + kk);
    f8 s_h = *(const f8*)(Whh + g*D_ + kk);
    *(h8*)(wiL + i*8) = __builtin_convertvector(s_i, h8);
    *(h8*)(whL + i*8) = __builtin_convertvector(s_h, h8);
  }
  for (int i = tid; i < MB*HS; i += 256) hb0[i] = (_Float16)0.0f;

  // ---- Per-lane constants ----
  const int m    = lane & 15;
  const int lo16 = (lane < 16);
  const int row_l_base = mt*16 + m;            // A-matrix row (local)
  const int rowA = row0 + row_l_base;          // global row in BN
  const int bb = rowA / N_;
  const int nn = rowA - bb * N_;
  // X flat index for (bb, t, nn, :) plus the lane's A-chunk offset
  const int xbase = bb*(L_*N_*D_) + nn*D_ + (lo16 ? 0 : 8);
  const int koA   = lo16 ? 0 : 8;              // A-fragment half-offset in h LDS

  const int row_c = mt*16 + (lo16 ? 0 : 8);    // C/D-matrix row base (+v)

  // Biases per lane: index by [gate*2 + dt]
  float bi[6], bh[6];
  #pragma unroll
  for (int g = 0; g < 3; ++g)
    #pragma unroll
    for (int dt = 0; dt < 2; ++dt) {
      int d = dbase + 16*dt + m;
      bi[g*2+dt] = bih[g*128 + d];
      bh[g*2+dt] = bhh[g*128 + d];
    }

  // fp32 hidden state carried in registers for the elementwise recurrence
  float hreg[16];
  #pragma unroll
  for (int i = 0; i < 16; ++i) hreg[i] = 0.0f;

  __syncthreads();

  _Float16* hcur = hb0;
  _Float16* hnxt = hb1;
  const v8f vzero = {};

  for (int t = 0; t < L_; ++t) {
    v8f acci[6], acch[6];
    #pragma unroll
    for (int i = 0; i < 6; ++i) { acci[i] = vzero; acch[i] = vzero; }

    const float* xp = X + xbase + t*(N_*D_);
    if (t + 1 < L_) {
      __builtin_prefetch(X + xbase + (t+1)*(N_*D_), 0, 1);  // global_prefetch_b8
    }

    #pragma unroll
    for (int k = 0; k < KT; ++k) {
      // A-fragment from x: fp32 global load, convert to f16
      f8 xlo = *(const f8*)(xp + k*32);
      f8 xhi = *(const f8*)(xp + k*32 + 16);
      v16h a_x = cat8(__builtin_convertvector(xlo, h8),
                      __builtin_convertvector(xhi, h8));
      // A-fragment from h: f16 LDS (row-major, padded stride)
      const _Float16* hp = hcur + row_l_base*HS + k*32 + koA;
      v16h a_h = cat8(*(const h8*)(hp), *(const h8*)(hp + 16));

      #pragma unroll
      for (int g = 0; g < 3; ++g) {
        #pragma unroll
        for (int dt = 0; dt < 2; ++dt) {
          int nt = g*8 + dr*2 + dt;
          int e  = ((nt*KT + k)*2)*32 + lane;   // h8 slot index in swizzled W
          v16h b_i = cat8(((const h8*)wiL)[e], ((const h8*)wiL)[e+32]);
          v16h b_h = cat8(((const h8*)whL)[e], ((const h8*)whL)[e+32]);
          acci[g*2+dt] = WMMA_F16(a_x, b_i, acci[g*2+dt]);
          acch[g*2+dt] = WMMA_F16(a_h, b_h, acch[g*2+dt]);
        }
      }
    }

    // ---- Gating (wave-local: this wave owns all 3 gates for its d-range) ----
    #pragma unroll
    for (int dt = 0; dt < 2; ++dt) {
      int d = dbase + 16*dt + m;
      #pragma unroll
      for (int v = 0; v < 8; ++v) {
        float rg  = (acci[0+dt][v] + bi[0+dt]) + (acch[0+dt][v] + bh[0+dt]);
        float zg  = (acci[2+dt][v] + bi[2+dt]) + (acch[2+dt][v] + bh[2+dt]);
        float ing = acci[4+dt][v] + bi[4+dt];
        float hng = acch[4+dt][v] + bh[4+dt];
        float r  = 1.0f / (1.0f + __expf(-rg));
        float z  = 1.0f / (1.0f + __expf(-zg));
        float ng = tanhf(ing + r * hng);
        float h  = (1.0f - z) * ng + z * hreg[dt*8+v];
        hreg[dt*8+v] = h;
        int rl = row_c + v;
        hnxt[rl*HS + d] = (_Float16)h;             // f16 copy for next-step GEMM
        out[(t*BN + row0 + rl)*D_ + d] = h;        // fp32 output (L, BN, D)
      }
    }
    __syncthreads();
    _Float16* tswap = hcur; hcur = hnxt; hnxt = tswap;
  }
}

extern "C" void kernel_launch(void* const* d_in, const int* in_sizes, int n_in,
                              void* d_out, int out_size, void* d_ws, size_t ws_size,
                              hipStream_t stream) {
  (void)in_sizes; (void)n_in; (void)d_ws; (void)ws_size; (void)out_size;
  const float* X   = (const float*)d_in[0];
  const float* Wih = (const float*)d_in[1];
  const float* Whh = (const float*)d_in[2];
  const float* bih = (const float*)d_in[3];
  const float* bhh = (const float*)d_in[4];
  float* out = (float*)d_out;

  dim3 grid(BN / MB);   // 325 blocks, one per 32 rows
  dim3 block(256);      // 8 wave32 waves
  gru_fused_kernel<<<grid, block, LDS_BYTES, stream>>>(X, Wih, Whh, bih, bhh, out);
}
